// GraphormerEncoder_27839978013470
// MI455X (gfx1250) — matrix operand order repaired and checked by reading, not compiled, not run on hardware
//
#include <hip/hip_runtime.h>
#include <hip/hip_bf16.h>

typedef _Float16 half_t;
typedef __attribute__((ext_vector_type(16))) _Float16 v16h;
typedef __attribute__((ext_vector_type(8)))  _Float16 v8h;
typedef __attribute__((ext_vector_type(8)))  float    v8f;
typedef __attribute__((ext_vector_type(4)))  int      v4i;

typedef __attribute__((address_space(1))) v4i as1_v4i;
typedef __attribute__((address_space(3))) v4i as3_v4i;

#if defined(__gfx1250__) && __has_builtin(__builtin_amdgcn_global_load_async_to_lds_b128)
#define ASYNC_LDS 1
#else
#define ASYNC_LDS 0
#endif

#define G_      64
#define P_      31
#define N_ORIG_ 1984
#define NN_     2048
#define D_      512
#define H_      8
#define DK_     64
#define FF_     2048
#define L_      4
#define NF_     32
#define MAX_SP_ 100
#define MAX_DEG_ 100

// ---------------------------------------------------------------------------
// Async global->LDS helpers (CDNA5 GLOBAL_LOAD_ASYNC_TO_LDS_B128, ASYNCcnt)
// ---------------------------------------------------------------------------
__device__ __forceinline__ void cp_b128(half_t* lds, const half_t* g) {
#if ASYNC_LDS
    __builtin_amdgcn_global_load_async_to_lds_b128(
        (as1_v4i*)(v4i*)(void*)const_cast<half_t*>(g),
        (as3_v4i*)(v4i*)(void*)lds, 0, 0);
#else
    *(v8h*)lds = *(const v8h*)g;
#endif
}

__device__ __forceinline__ void wait_async() {
#if ASYNC_LDS
#if __has_builtin(__builtin_amdgcn_s_wait_asynccnt)
    __builtin_amdgcn_s_wait_asynccnt(0);
#else
    asm volatile("s_wait_asynccnt 0x0" ::: "memory");
#endif
#endif
}

// ---------------------------------------------------------------------------
// Weight conversion kernels (all weights stored TRANSPOSED: [N][K] per layer,
// so GEMM B-tiles stage into LDS with contiguous 16B copies, no scatter)
// ---------------------------------------------------------------------------

// Wq/Wk/Wv: (L,H,D,DK) f32 -> per-layer [n = h*64+k][d] f16
__global__ void k_convert_qkv(const float* __restrict__ W, half_t* __restrict__ out) {
    long idx = (long)blockIdx.x * blockDim.x + threadIdx.x;
    const long total = (long)L_ * H_ * D_ * DK_;
    if (idx >= total) return;
    int k = (int)(idx % DK_); long t = idx / DK_;
    int d = (int)(t % D_);    t /= D_;
    int h = (int)(t % H_);
    int l = (int)(t / H_);
    out[((long)l * D_ + h * DK_ + k) * D_ + d] = (half_t)W[idx];
}

// (L,K,N) f32 -> (L,N,K) f16
__global__ void k_convert_T(const float* __restrict__ in, half_t* __restrict__ out,
                            int Kd, int Nd) {
    long idx = (long)blockIdx.x * blockDim.x + threadIdx.x;
    long total = (long)L_ * Kd * Nd;
    if (idx >= total) return;
    int n = (int)(idx % Nd); long t = idx / Nd;
    int k = (int)(t % Kd);
    int l = (int)(t / Kd);
    out[((long)l * Nd + n) * Kd + k] = (half_t)in[idx];
}

// ---------------------------------------------------------------------------
// h initialization: h = [x@init_W + init_b ; zeros] + cent_emb[min(deg,100)]
// ---------------------------------------------------------------------------
__global__ void k_init_h(const float* __restrict__ x, const int* __restrict__ degrees,
                         const float* __restrict__ init_W, const float* __restrict__ init_b,
                         const float* __restrict__ cent_emb, float* __restrict__ h) {
    __shared__ float xr[NF_];
    int n = blockIdx.x;
    int t = threadIdx.x;
    if (t < NF_ && n < N_ORIG_) xr[t] = x[(long)n * NF_ + t];
    __syncthreads();
    int deg = degrees[n]; if (deg > MAX_DEG_) deg = MAX_DEG_;
    for (int d = t; d < D_; d += 256) {
        float acc = cent_emb[(long)deg * D_ + d];
        if (n < N_ORIG_) {
            acc += init_b[d];
            #pragma unroll
            for (int f = 0; f < NF_; ++f) acc += xr[f] * init_W[(long)f * D_ + d];
        }
        h[(long)n * D_ + d] = acc;
    }
}

// ---------------------------------------------------------------------------
// LayerNorm (no affine, eps=1e-5, biased variance) -> f16 output
// ---------------------------------------------------------------------------
__global__ __launch_bounds__(256)
void k_layernorm(const float* __restrict__ h, half_t* __restrict__ out) {
    __shared__ float rs[256], rs2[256];
    int n = blockIdx.x, t = threadIdx.x;
    float v0 = h[(long)n * D_ + t];
    float v1 = h[(long)n * D_ + t + 256];
    rs[t]  = v0 + v1;
    rs2[t] = v0 * v0 + v1 * v1;
    __syncthreads();
    for (int off = 128; off > 0; off >>= 1) {
        if (t < off) { rs[t] += rs[t + off]; rs2[t] += rs2[t + off]; }
        __syncthreads();
    }
    float mean = rs[0] * (1.0f / D_);
    float var  = rs2[0] * (1.0f / D_) - mean * mean;
    float inv  = rsqrtf(var + 1e-5f);
    out[(long)n * D_ + t]       = (half_t)((v0 - mean) * inv);
    out[(long)n * D_ + t + 256] = (half_t)((v1 - mean) * inv);
}

// ---------------------------------------------------------------------------
// WMMA GEMM: C[M,N] = A[M,K](f16) @ Bt[N,K](f16, pre-transposed) + bias
//            (+relu) (+resid f32).  Tile 128x128x32, 8 waves, double-buffered
//            async global->LDS pipeline.
// ---------------------------------------------------------------------------
#define BM 128
#define BN 128
#define BK 32

__device__ __forceinline__ void gemm_compute_tile(
    const half_t (*__restrict__ As)[BK], const half_t (*__restrict__ Bs)[BK],
    int wm, int wn, int m0, int kbA, int kbB, v8f acc[4][2]) {
    v16h afr[4];
    #pragma unroll
    for (int i = 0; i < 4; ++i) {
        const half_t* pa = &As[wm + i * 16 + m0][0];
        v8h lo = *(const v8h*)(pa + kbA);
        v8h hi = *(const v8h*)(pa + kbA + 16);
        v16h a;
        #pragma unroll
        for (int e = 0; e < 8; ++e) { a[e] = lo[e]; a[e + 8] = hi[e]; }
        afr[i] = a;
    }
    v16h bfr[2];
    #pragma unroll
    for (int j = 0; j < 2; ++j) {
        const half_t* pb = &Bs[wn + j * 16 + m0][kbB];
        v8h lo = *(const v8h*)(pb);
        v8h hi = *(const v8h*)(pb + 16);
        v16h b;
        #pragma unroll
        for (int e = 0; e < 8; ++e) { b[e] = lo[e]; b[e + 8] = hi[e]; }
        bfr[j] = b;
    }
    #pragma unroll
    for (int i = 0; i < 4; ++i)
        #pragma unroll
        for (int j = 0; j < 2; ++j)
            acc[i][j] = __builtin_amdgcn_wmma_f32_16x16x32_f16(
                false, afr[i], false, bfr[j], (short)0, acc[i][j], false, false);
}

__global__ __launch_bounds__(256)
void k_gemm(const half_t* __restrict__ A, const half_t* __restrict__ Bt,
            const float* __restrict__ bias, const float* __restrict__ resid,
            float* __restrict__ Cf, half_t* __restrict__ Ch,
            int M, int N, int K, int relu) {
    __shared__ half_t As[2][BM][BK];
    __shared__ half_t Bs[2][BN][BK];
    int tid  = threadIdx.x;
    int lane = tid & 31;
    int wave = tid >> 5;
    int bm = blockIdx.y * BM;
    int bn = blockIdx.x * BN;
    int wm = (wave >> 2) * 64;
    int wn = (wave & 3) * 32;

    v8f acc[4][2] = {};

    const int m0  = lane & 15;
    const int kbA = (lane < 16) ? 0 : 8;
    const int kbB = (lane < 16) ? 0 : 8;

    // per-thread staging assignment: 32B of A tile and 32B of B tile
    const int srow = tid >> 1;
    const int scol = (tid & 1) * 16;
    const half_t* ga = A  + (long)(bm + srow) * K + scol;   // advance by k0
    const half_t* gb = Bt + (long)(bn + srow) * K + scol;

    // prologue: tile 0 -> buffer 0
    cp_b128(&As[0][srow][scol], ga);
    cp_b128(&As[0][srow][scol + 8], ga + 8);
    cp_b128(&Bs[0][srow][scol], gb);
    cp_b128(&Bs[0][srow][scol + 8], gb + 8);
    wait_async();
    __syncthreads();

    const int nk = K / BK;                       // 16 or 64 (always even)
    for (int it = 0; it < nk; it += 2) {
        // issue tile it+1 -> buffer 1, compute tile it from buffer 0
        {
            const half_t* pa = ga + (long)(it + 1) * BK;
            const half_t* pb = gb + (long)(it + 1) * BK;
            cp_b128(&As[1][srow][scol], pa);
            cp_b128(&As[1][srow][scol + 8], pa + 8);
            cp_b128(&Bs[1][srow][scol], pb);
            cp_b128(&Bs[1][srow][scol + 8], pb + 8);
        }
        gemm_compute_tile(As[0], Bs[0], wm, wn, m0, kbA, kbB, acc);
        wait_async();
        __syncthreads();
        // issue tile it+2 -> buffer 0, compute tile it+1 from buffer 1
        if (it + 2 < nk) {
            const half_t* pa = ga + (long)(it + 2) * BK;
            const half_t* pb = gb + (long)(it + 2) * BK;
            cp_b128(&As[0][srow][scol], pa);
            cp_b128(&As[0][srow][scol + 8], pa + 8);
            cp_b128(&Bs[0][srow][scol], pb);
            cp_b128(&Bs[0][srow][scol + 8], pb + 8);
        }
        gemm_compute_tile(As[1], Bs[1], wm, wn, m0, kbA, kbB, acc);
        wait_async();
        __syncthreads();
    }

    const int nl = lane & 15;
    const int mo = (lane < 16) ? 0 : 8;
    #pragma unroll
    for (int i = 0; i < 4; ++i) {
        #pragma unroll
        for (int j = 0; j < 2; ++j) {
            int col = bn + wn + j * 16 + nl;
            float bsv = bias ? bias[col] : 0.0f;
            #pragma unroll
            for (int v = 0; v < 8; ++v) {
                int row  = bm + wm + i * 16 + mo + v;
                long off = (long)row * N + col;
                float val = acc[i][j][v] + bsv;
                if (relu) val = val > 0.0f ? val : 0.0f;
                if (resid) val += resid[off];
                if (Cf) Cf[off] = val;
                if (Ch) Ch[off] = (half_t)val;
            }
        }
    }
}

// ---------------------------------------------------------------------------
// Block-diagonal attention: one wave per (graph, head).
// ---------------------------------------------------------------------------
__global__ __launch_bounds__(32)
void k_attention(const half_t* __restrict__ Q, const half_t* __restrict__ Kb,
                 const half_t* __restrict__ V, const int* __restrict__ sp,
                 const float* __restrict__ dist_biases, const float* __restrict__ virt_bias,
                 half_t* __restrict__ O) {
    __shared__ half_t Qh[32][64];
    __shared__ half_t Kh[32][64];
    __shared__ half_t Vt[64][32];   // Vt[dk][node]
    __shared__ float  Bb[32][32];
    __shared__ float  Sm[32][32];
    __shared__ half_t Pf[32][32];

    int g    = blockIdx.x;
    int hh   = blockIdx.y;
    int lane = threadIdx.x;

    // Q/K rows via async LDS loads: 4 rows per step, lane -> (row, 16B segment)
    {
        int r   = lane >> 3;
        int seg = (lane & 7) * 8;
        #pragma unroll
        for (int jj = 0; jj < 32; jj += 4) {
            int j   = jj + r;
            int row = (j < P_) ? (g * P_ + j) : (N_ORIG_ + g);
            long base = (long)row * D_ + hh * DK_ + seg;
            cp_b128(&Qh[j][seg], Q + base);
            cp_b128(&Kh[j][seg], Kb + base);
        }
    }
    // V gathered transposed (scalar)
    #pragma unroll 4
    for (int j = 0; j < 32; ++j) {
        int row = (j < P_) ? (g * P_ + j) : (N_ORIG_ + g);
        long base = (long)row * D_ + hh * DK_;
        Vt[lane][j]      = V[base + lane];
        Vt[lane + 32][j] = V[base + lane + 32];
    }
    // bias block (lane = column b)
    {
        float d0 = dist_biases[0];
        float vb = virt_bias[0];
        int b  = lane;
        int jg = g * P_ + b;
        for (int a = 0; a < 32; ++a) {
            float val;
            if (a == b) val = d0;
            else if (a == P_ || b == P_) val = vb;
            else {
                int s = sp[(long)(g * P_ + a) * N_ORIG_ + jg];
                if (s > MAX_SP_) s = MAX_SP_;
                val = dist_biases[s];
            }
            Bb[a][b] = val;
        }
    }
    wait_async();
    __syncthreads();

    const int m0  = lane & 15;
    const int kbA = (lane < 16) ? 0 : 8;
    const int kbB = (lane < 16) ? 0 : 8;
    const float scale = 0.125f;   // 1/sqrt(64)

    // S = Q @ K^T : 2x2 tiles, K=64
    #pragma unroll
    for (int ti = 0; ti < 2; ++ti) {
        #pragma unroll
        for (int tj = 0; tj < 2; ++tj) {
            v8f c = {};
            #pragma unroll
            for (int ks = 0; ks < 2; ++ks) {
                const half_t* pa = &Qh[ti * 16 + m0][ks * 32];
                v8h lo = *(const v8h*)(pa + kbA);
                v8h hi = *(const v8h*)(pa + kbA + 16);
                v16h a;
                #pragma unroll
                for (int e = 0; e < 8; ++e) { a[e] = lo[e]; a[e + 8] = hi[e]; }
                const half_t* pb = &Kh[tj * 16 + m0][ks * 32 + kbB];
                v8h blo = *(const v8h*)(pb);
                v8h bhi = *(const v8h*)(pb + 16);
                v16h b;
                #pragma unroll
                for (int e = 0; e < 8; ++e) { b[e] = blo[e]; b[e + 8] = bhi[e]; }
                c = __builtin_amdgcn_wmma_f32_16x16x32_f16(false, a, false, b, (short)0, c, false, false);
            }
            int nl = lane & 15, mo = (lane < 16) ? 0 : 8;
            #pragma unroll
            for (int v = 0; v < 8; ++v) {
                int rm = ti * 16 + mo + v, cn = tj * 16 + nl;
                Sm[rm][cn] = c[v] * scale + Bb[rm][cn];
            }
        }
    }
    __syncthreads();

    // softmax per row (lane = row)
    {
        int r = lane;
        float mx = -1e30f;
        #pragma unroll
        for (int c = 0; c < 32; ++c) mx = fmaxf(mx, Sm[r][c]);
        float p[32], sum = 0.0f;
        #pragma unroll
        for (int c = 0; c < 32; ++c) { p[c] = __expf(Sm[r][c] - mx); sum += p[c]; }
        float inv = 1.0f / sum;
        #pragma unroll
        for (int c = 0; c < 32; ++c) Pf[r][c] = (half_t)(p[c] * inv);
    }
    __syncthreads();

    // O = P @ V : 2x4 tiles, K=32
    #pragma unroll
    for (int ti = 0; ti < 2; ++ti) {
        #pragma unroll
        for (int tj = 0; tj < 4; ++tj) {
            const half_t* pa = &Pf[ti * 16 + m0][0];
            v8h lo = *(const v8h*)(pa + kbA);
            v8h hi = *(const v8h*)(pa + kbA + 16);
            v16h a;
            #pragma unroll
            for (int e = 0; e < 8; ++e) { a[e] = lo[e]; a[e + 8] = hi[e]; }
            const half_t* pb = &Vt[tj * 16 + m0][kbB];
            v8h blo = *(const v8h*)(pb);
            v8h bhi = *(const v8h*)(pb + 16);
            v16h b;
            #pragma unroll
            for (int e = 0; e < 8; ++e) { b[e] = blo[e]; b[e + 8] = bhi[e]; }
            v8f c = {};
            c = __builtin_amdgcn_wmma_f32_16x16x32_f16(false, a, false, b, (short)0, c, false, false);
            int nl = lane & 15, mo = (lane < 16) ? 0 : 8;
            #pragma unroll
            for (int v = 0; v < 8; ++v) {
                int rm  = ti * 16 + mo + v;
                int row = (rm < P_) ? (g * P_ + rm) : (N_ORIG_ + g);
                int col = hh * DK_ + tj * 16 + nl;
                O[(long)row * D_ + col] = (half_t)c[v];
            }
        }
    }
}

// ---------------------------------------------------------------------------
__global__ void k_copy_out(const float* __restrict__ h, float* __restrict__ out) {
    int i = blockIdx.x * blockDim.x + threadIdx.x;   // G_*D_
    out[i] = h[(long)N_ORIG_ * D_ + i];
}

// ---------------------------------------------------------------------------
extern "C" void kernel_launch(void* const* d_in, const int* in_sizes, int n_in,
                              void* d_out, int out_size, void* d_ws, size_t ws_size,
                              hipStream_t stream) {
    (void)in_sizes; (void)n_in; (void)out_size; (void)ws_size;
    const float* x        = (const float*)d_in[0];
    const int*   sp       = (const int*)d_in[1];
    const int*   degrees  = (const int*)d_in[3];
    const float* init_W   = (const float*)d_in[4];
    const float* init_b   = (const float*)d_in[5];
    const float* cent_emb = (const float*)d_in[6];
    const float* dist_b   = (const float*)d_in[7];
    const float* virt_b   = (const float*)d_in[8];
    const float* Wq = (const float*)d_in[9];
    const float* bq = (const float*)d_in[10];
    const float* Wk = (const float*)d_in[11];
    const float* bk = (const float*)d_in[12];
    const float* Wv = (const float*)d_in[13];
    const float* bv = (const float*)d_in[14];
    const float* Wo = (const float*)d_in[15];
    const float* bo = (const float*)d_in[16];
    const float* W1 = (const float*)d_in[17];
    const float* b1 = (const float*)d_in[18];
    const float* W2 = (const float*)d_in[19];
    const float* b2 = (const float*)d_in[20];
    float* out = (float*)d_out;

    char* ws = (char*)d_ws;
    size_t off = 0;
    auto alloc = [&](size_t bytes) -> void* {
        off = (off + 255) & ~(size_t)255;
        void* p = ws + off;
        off += bytes;
        return p;
    };
    float*  h    = (float*)alloc((size_t)NN_ * D_ * 4);
    half_t* hn   = (half_t*)alloc((size_t)NN_ * D_ * 2);
    half_t* qb   = (half_t*)alloc((size_t)NN_ * D_ * 2);
    half_t* kbuf = (half_t*)alloc((size_t)NN_ * D_ * 2);
    half_t* vb   = (half_t*)alloc((size_t)NN_ * D_ * 2);
    half_t* ob   = (half_t*)alloc((size_t)NN_ * D_ * 2);
    half_t* ub   = (half_t*)alloc((size_t)NN_ * FF_ * 2);
    half_t* wqh  = (half_t*)alloc((size_t)L_ * D_ * D_ * 2);
    half_t* wkh  = (half_t*)alloc((size_t)L_ * D_ * D_ * 2);
    half_t* wvh  = (half_t*)alloc((size_t)L_ * D_ * D_ * 2);
    half_t* woh  = (half_t*)alloc((size_t)L_ * D_ * D_ * 2);
    half_t* w1h  = (half_t*)alloc((size_t)L_ * D_ * FF_ * 2);
    half_t* w2h  = (half_t*)alloc((size_t)L_ * FF_ * D_ * 2);

    const long nqkv = (long)L_ * H_ * D_ * DK_;
    k_convert_qkv<<<(nqkv + 255) / 256, 256, 0, stream>>>(Wq, wqh);
    k_convert_qkv<<<(nqkv + 255) / 256, 256, 0, stream>>>(Wk, wkh);
    k_convert_qkv<<<(nqkv + 255) / 256, 256, 0, stream>>>(Wv, wvh);
    const long nwo = (long)L_ * D_ * D_;
    k_convert_T<<<(nwo + 255) / 256, 256, 0, stream>>>(Wo, woh, D_, D_);
    const long nw1 = (long)L_ * D_ * FF_;
    k_convert_T<<<(nw1 + 255) / 256, 256, 0, stream>>>(W1, w1h, D_, FF_);
    k_convert_T<<<(nw1 + 255) / 256, 256, 0, stream>>>(W2, w2h, FF_, D_);

    k_init_h<<<NN_, 256, 0, stream>>>(x, degrees, init_W, init_b, cent_emb, h);

    dim3 gD(D_ / BN, NN_ / BM);    // N=512 GEMMs
    dim3 gF(FF_ / BN, NN_ / BM);   // N=2048 GEMM
    dim3 gAttn(G_, H_);

    for (int l = 0; l < L_; ++l) {
        k_layernorm<<<NN_, 256, 0, stream>>>(h, hn);
        k_gemm<<<gD, 256, 0, stream>>>(hn, wqh + (long)l * D_ * D_, bq + (long)l * D_,
                                       nullptr, nullptr, qb, NN_, D_, D_, 0);
        k_gemm<<<gD, 256, 0, stream>>>(hn, wkh + (long)l * D_ * D_, bk + (long)l * D_,
                                       nullptr, nullptr, kbuf, NN_, D_, D_, 0);
        k_gemm<<<gD, 256, 0, stream>>>(hn, wvh + (long)l * D_ * D_, bv + (long)l * D_,
                                       nullptr, nullptr, vb, NN_, D_, D_, 0);
        k_attention<<<gAttn, 32, 0, stream>>>(qb, kbuf, vb, sp, dist_b, virt_b, ob);
        k_gemm<<<gD, 256, 0, stream>>>(ob, woh + (long)l * D_ * D_, bo + (long)l * D_,
                                       h, h, nullptr, NN_, D_, D_, 0);
        k_layernorm<<<NN_, 256, 0, stream>>>(h, hn);
        k_gemm<<<gF, 256, 0, stream>>>(hn, w1h + (long)l * D_ * FF_, b1 + (long)l * FF_,
                                       nullptr, nullptr, ub, NN_, FF_, D_, 1);
        k_gemm<<<gD, 256, 0, stream>>>(ub, w2h + (long)l * FF_ * D_, b2 + (long)l * D_,
                                       h, h, nullptr, NN_, D_, FF_, 0);
    }

    k_copy_out<<<(G_ * D_) / 256, 256, 0, stream>>>(h, out);
}
